// SimpleGATLayer_79877801771537
// MI455X (gfx1250) — matrix operand order repaired and checked by reading, not compiled, not run on hardware
//
#include <hip/hip_runtime.h>

#define BB   16
#define NN   1024
#define DIN  256
#define DOUT 256
#define NEG_INF (-9.0e15f)

typedef __attribute__((ext_vector_type(8)))  float  v8f;
typedef __attribute__((ext_vector_type(2)))  __bf16 v2bf;
typedef __attribute__((ext_vector_type(16))) __bf16 v16bf;

union Frag16 { v16bf v; unsigned u[8]; };

#if defined(__has_builtin)
#if __has_builtin(__builtin_amdgcn_cvt_pk_bf16_f32)
#define HAVE_CVT_PK_BF16 1
#endif
#endif

// round-to-nearest-even f32 -> bf16 (bit pattern)
static __device__ __forceinline__ unsigned short f2bf(float f) {
  unsigned u = __float_as_uint(f);
  u = (u + 0x7FFFu + ((u >> 16) & 1u)) >> 16;
  return (unsigned short)u;
}
static __device__ __forceinline__ unsigned pack_bf2(float lo, float hi) {
#ifdef HAVE_CVT_PK_BF16
  v2bf p = __builtin_amdgcn_cvt_pk_bf16_f32(lo, hi);
  return __builtin_bit_cast(unsigned, p);
#else
  return (unsigned)f2bf(lo) | ((unsigned)f2bf(hi) << 16);
#endif
}

static __device__ __forceinline__ v8f wmma_bf16(v16bf a, v16bf b, v8f c) {
  // (neg_a, A, neg_b, B, c_mod, C, reuse_a, reuse_b)
  return __builtin_amdgcn_wmma_f32_16x16x32_bf16(false, a, false, b, (short)0, c, false, false);
}

// ---------------------------------------------------------------------------
// Kernel 0a: x (f32) -> xb (bf16, packed pairs along k).  One u32 per thread.
// ---------------------------------------------------------------------------
__global__ __launch_bounds__(256) void cvt_x(const float* __restrict__ x,
                                             unsigned* __restrict__ xb) {
  const int i = blockIdx.x * 256 + threadIdx.x;     // pair index
  const float2 v = *(const float2*)(x + 2 * (size_t)i);
  xb[i] = pack_bf2(v.x, v.y);
}

// ---------------------------------------------------------------------------
// Kernel 0b: W[k][n] (f32) -> wt[n][k] (bf16, packed pairs along k).
// ---------------------------------------------------------------------------
__global__ __launch_bounds__(256) void cvt_w(const float* __restrict__ W,
                                             unsigned* __restrict__ wt) {
  const int i  = blockIdx.x * 256 + threadIdx.x;    // in [0, DOUT*DIN/2)
  const int n  = i / (DIN / 2);
  const int kp = i % (DIN / 2);
  wt[i] = pack_bf2(W[(size_t)(2 * kp) * DOUT + n], W[(size_t)(2 * kp + 1) * DOUT + n]);
}

// ---------------------------------------------------------------------------
// Kernel 1: h = x @ W   (M=B*N=16384, K=256, N=256), bf16 WMMA, f32 accum.
// Inner loop is pure b128 loads of pre-packed bf16 pairs + WMMA.
// Writes h (f32, row-major) and h^T per batch (bf16) for the second GEMM.
// ---------------------------------------------------------------------------
__global__ __launch_bounds__(256) void gat_gemm_h(const unsigned* __restrict__ xb,
                                                  const unsigned* __restrict__ wt,
                                                  float* __restrict__ h,
                                                  unsigned short* __restrict__ ht) {
  const int lane = threadIdx.x & 31;
  const int wave = threadIdx.x >> 5;
  const int half = lane >> 4;     // 0: lanes 0-15, 1: lanes 16-31
  const int mr   = lane & 15;
  const int i0   = blockIdx.x * 16;          // global row tile over B*N rows
  const int n0a  = wave * 32;
  const int n0b  = wave * 32 + 16;

  const unsigned* xrow = xb + (size_t)(i0 + mr) * (DIN / 2);
  const unsigned* wca  = wt + (size_t)(n0a + mr) * (DIN / 2);
  const unsigned* wcb  = wt + (size_t)(n0b + mr) * (DIN / 2);

  v8f acc0 = {}; v8f acc1 = {};

#pragma unroll 2
  for (int ks = 0; ks < DIN; ks += 32) {
    // A fragment: 16x32 bf16. VGPR v: k = 16*half + 2v + {0,1}
    Frag16 A;
#pragma unroll
    for (int v = 0; v < 8; ++v) {
      const int k = ks + (half << 4) + (v << 1);
      A.u[v] = xrow[k >> 1];
    }
    // B fragment: 32x16 bf16. VGPR r: n = mr, k = 16*half + 2r + {0,1}
    Frag16 B0, B1;
#pragma unroll
    for (int r = 0; r < 8; ++r) {
      const int k = ks + (half << 4) + (r << 1);
      B0.u[r] = wca[k >> 1];
      B1.u[r] = wcb[k >> 1];
    }
    acc0 = wmma_bf16(A.v, B0.v, acc0);
    acc1 = wmma_bf16(A.v, B1.v, acc1);
  }

  // C layout: VGPR r holds row (r + 8*half), col = mr (within tile)
  const int b     = i0 >> 10;       // batch (tiles never straddle batches)
  const int jrow0 = i0 & (NN - 1);  // row within batch
#pragma unroll
  for (int r = 0; r < 8; ++r) {
    const int row = r + (half << 3);
    const float v0 = acc0[r];
    const float v1 = acc1[r];
    h[(size_t)(i0 + row) * DOUT + n0a + mr] = v0;
    h[(size_t)(i0 + row) * DOUT + n0b + mr] = v1;
    ht[((size_t)b * DOUT + n0a + mr) * NN + jrow0 + row] = f2bf(v0);
    ht[((size_t)b * DOUT + n0b + mr) * NN + jrow0 + row] = f2bf(v1);
  }
}

// ---------------------------------------------------------------------------
// Kernel 2: src = h @ a1, dst = h @ a2.  One wave per row, shuffle-reduce.
// ---------------------------------------------------------------------------
__global__ __launch_bounds__(256) void gat_src_dst(const float* __restrict__ h,
                                                   const float* __restrict__ a,
                                                   float* __restrict__ src,
                                                   float* __restrict__ dst) {
  const int lane = threadIdx.x & 31;
  const int row  = blockIdx.x * 8 + (threadIdx.x >> 5);
  if (row >= BB * NN) return;
  float s = 0.f, t = 0.f;
  for (int c = lane; c < DOUT; c += 32) {
    const float hv = h[(size_t)row * DOUT + c];
    s += hv * a[c];
    t += hv * a[DOUT + c];
  }
#pragma unroll
  for (int off = 16; off; off >>= 1) {
    s += __shfl_xor(s, off, 32);
    t += __shfl_xor(t, off, 32);
  }
  if (lane == 0) { src[row] = s; dst[row] = t; }
}

// ---------------------------------------------------------------------------
// Kernel 3: fused masked-softmax attention: out = softmax(mask(lrelu(e))) @ h.
// Block = 8 waves = 64 i-rows x 256 d-cols:
//   wave w: i-tile (w&3), column half (w>>2) -> 8 WMMA accumulator tiles.
// Per 32-wide j-step: the 256x32 h^T tile is DMA'd into LDS with
// GLOBAL_LOAD_ASYNC_TO_LDS_B128 (ASYNCcnt), overlapping the whole score /
// online-softmax computation; s_wait_asynccnt 0 + barrier before the 8 WMMAs
// whose B fragments come from LDS as contiguous b128 reads. Double-buffered.
// ---------------------------------------------------------------------------
__global__ __launch_bounds__(256) void gat_attn(const float* __restrict__ adj,
                                                const unsigned short* __restrict__ ht,
                                                const float* __restrict__ src,
                                                const float* __restrict__ dst,
                                                float* __restrict__ out) {
  __shared__ unsigned char sm[2][DOUT * 32 * 2];   // 2 x 16 KB (bf16 tile [d][j])

  const int tid  = threadIdx.x;
  const int lane = tid & 31;
  const int wave = tid >> 5;
  const int half = lane >> 4;
  const int mr   = lane & 15;

  const int b    = blockIdx.x >> 4;           // 16 blocks of 64 rows per batch
  const int i0   = (blockIdx.x & 15) * 64;    // block's first row in batch
  const int it   = wave & 3;                  // wave's i-tile within block
  const int ch   = wave >> 2;                 // column half: 0 or 1
  const int ib   = i0 + it * 16;              // wave's first row
  const int d0   = ch * 128;                  // wave's first column

  const float  s_i    = src[b * NN + ib + mr];
  const float* adjrow = adj + ((size_t)b * NN + ib + mr) * NN;
  const float* dstb   = dst + b * NN;
  // staging source: this thread copies row d = tid of h^T (32 bf16 per step)
  const unsigned short* htrow = ht + ((size_t)b * DOUT + tid) * NN;
  // LDS byte offset of this thread's staging row (generic LDS pointers carry
  // the DS offset in their low 32 bits)
  const unsigned lds_row0 = (unsigned)(uintptr_t)(&sm[0][0]) + (unsigned)tid * 64u;

  float mrun = NEG_INF;   // init to reference's mask value: fully-masked rows
  float lsum = 0.f;       // reproduces reference's uniform softmax exactly
  v8f acc[8];
#pragma unroll
  for (int t = 0; t < 8; ++t) acc[t] = (v8f){};

  for (int j0 = 0; j0 < NN; j0 += 32) {
    const int s = (j0 >> 5) & 1;
    const unsigned char* buf = sm[s];
    // ---- async DMA: h^T[d = tid][j0..j0+32) -> LDS (64 B per lane) ----
    {
      const unsigned lds_dst = lds_row0 + (unsigned)s * (DOUT * 32 * 2);
      const unsigned short* gsrc = htrow + j0;
      asm volatile(
          "global_load_async_to_lds_b128 %0, %1, off\n\t"
          "global_load_async_to_lds_b128 %0, %1, off offset:16\n\t"
          "global_load_async_to_lds_b128 %0, %1, off offset:32\n\t"
          "global_load_async_to_lds_b128 %0, %1, off offset:48"
          :: "v"(lds_dst), "v"(gsrc) : "memory");
    }
    // prefetch next adj tile of this row (the only HBM-bound stream)
    if (j0 + 32 < NN) {
      __builtin_prefetch(adjrow + j0 + 32, 0, 1);
    }
    // ---- score tile in A-fragment layout (row = mr, 16 k's per lane) ----
    float e[16];
    float lmax = NEG_INF;
#pragma unroll
    for (int v = 0; v < 8; ++v) {
      const int k = (half << 4) + (v << 1);
      const float2 dj = *(const float2*)(dstb  + j0 + k);
      const float2 aj = *(const float2*)(adjrow + j0 + k);
      float e0 = s_i + dj.x; e0 = (e0 >= 0.f) ? e0 : 0.2f * e0; e0 = (aj.x > 0.f) ? e0 : NEG_INF;
      float e1 = s_i + dj.y; e1 = (e1 >= 0.f) ? e1 : 0.2f * e1; e1 = (aj.y > 0.f) ? e1 : NEG_INF;
      e[2 * v] = e0; e[2 * v + 1] = e1;
      lmax = fmaxf(lmax, fmaxf(e0, e1));
    }
    lmax = fmaxf(lmax, __shfl_xor(lmax, 16, 32));     // combine lane halves
    const float mnew  = fmaxf(mrun, lmax);
    const float scale = __expf(mrun - mnew);

    float psum = 0.f;
    Frag16 P;
#pragma unroll
    for (int v = 0; v < 8; ++v) {
      const float p0 = __expf(e[2 * v]     - mnew);
      const float p1 = __expf(e[2 * v + 1] - mnew);
      psum += p0 + p1;
      P.u[v] = pack_bf2(p0, p1);
    }
    psum += __shfl_xor(psum, 16, 32);
    lsum = lsum * scale + psum;
    mrun = mnew;

    // rescale accumulators: C VGPR r holds row (r + 8*half); fetch that
    // row's scale from lane (r + 8*half) via ds_bpermute.
    float sc[8];
#pragma unroll
    for (int r = 0; r < 8; ++r) sc[r] = __shfl(scale, r + (half << 3), 32);
#pragma unroll
    for (int t = 0; t < 8; ++t)
#pragma unroll
      for (int r = 0; r < 8; ++r) acc[t][r] *= sc[r];

    // async transfers done (this wave), then make tile visible to all waves
    asm volatile("s_wait_asynccnt 0" ::: "memory");
    __syncthreads();

    // ---- 8 WMMAs: B fragments from LDS (32 contiguous bytes per lane) ----
#pragma unroll
    for (int t = 0; t < 8; ++t) {
      const int d = d0 + t * 16 + mr;                 // this lane's column
      const uint4* l = (const uint4*)(buf + d * 64 + half * 32);
      Frag16 Bf;
      const uint4 q0 = l[0];
      Bf.u[0] = q0.x; Bf.u[1] = q0.y; Bf.u[2] = q0.z; Bf.u[3] = q0.w;
      const uint4 q1 = l[1];
      Bf.u[4] = q1.x; Bf.u[5] = q1.y; Bf.u[6] = q1.z; Bf.u[7] = q1.w;
      acc[t] = wmma_bf16(P.v, Bf.v, acc[t]);
    }
  }

  // normalize and store (out is f32)
  float linv[8];
#pragma unroll
  for (int r = 0; r < 8; ++r) linv[r] = 1.f / __shfl(lsum, r + (half << 3), 32);
#pragma unroll
  for (int t = 0; t < 8; ++t) {
#pragma unroll
    for (int r = 0; r < 8; ++r) {
      const int row = ib + r + (half << 3);
      out[((size_t)(b * NN) + row) * DOUT + d0 + t * 16 + mr] = acc[t][r] * linv[r];
    }
  }
}

// ---------------------------------------------------------------------------
extern "C" void kernel_launch(void* const* d_in, const int* in_sizes, int n_in,
                              void* d_out, int out_size, void* d_ws, size_t ws_size,
                              hipStream_t stream) {
  const float* x   = (const float*)d_in[0];   // (B,N,DIN)
  const float* adj = (const float*)d_in[1];   // (B,N,N)
  const float* W   = (const float*)d_in[2];   // (DIN,DOUT)
  const float* a   = (const float*)d_in[3];   // (2*DOUT,1)
  float* out = (float*)d_out;                 // (B,N,DOUT) f32

  char* ws = (char*)d_ws;
  const size_t OFF_H   = 0;                              // 16 MB
  const size_t OFF_HT  = (size_t)16 << 20;               // 8 MB
  const size_t OFF_SRC = (size_t)24 << 20;               // 64 KB
  const size_t OFF_DST = OFF_SRC + 65536;                // 64 KB
  const size_t OFF_XB  = OFF_DST + 65536;                // 8 MB
  const size_t OFF_WT  = OFF_XB + ((size_t)8 << 20);     // 128 KB

  float*          h   = (float*)(ws + OFF_H);
  unsigned short* ht  = (unsigned short*)(ws + OFF_HT);
  float*          src = (float*)(ws + OFF_SRC);
  float*          dst = (float*)(ws + OFF_DST);
  unsigned*       xbp = (unsigned*)(ws + OFF_XB);
  unsigned*       wtp = (unsigned*)(ws + OFF_WT);

  cvt_x<<<(BB * NN * DIN / 2) / 256, 256, 0, stream>>>(x, xbp);
  cvt_w<<<(DOUT * DIN / 2) / 256, 256, 0, stream>>>(W, wtp);
  gat_gemm_h<<<(BB * NN) / 16, 256, 0, stream>>>(xbp, wtp, h, ht);
  gat_src_dst<<<(BB * NN) / 8, 256, 0, stream>>>(h, a, src, dst);
  gat_attn<<<BB * (NN / 64), 256, 0, stream>>>(adj, ht, src, dst, out);
}
